// Decoder_23519240913589
// MI455X (gfx1250) — matrix operand order repaired and checked by reading, not compiled, not run on hardware
//
#include <hip/hip_runtime.h>

// ---------------- problem constants ----------------
#define B_ 8
#define H_ 8
#define I_ 1024
#define J_ 1024
#define EPSLN 1e-3f

// ---------------- types ----------------
typedef __bf16 v16bf __attribute__((ext_vector_type(16)));
typedef __bf16 bf16x4 __attribute__((ext_vector_type(4)));
typedef float  v8f   __attribute__((ext_vector_type(8)));
typedef unsigned int u32x4 __attribute__((ext_vector_type(4)));
typedef unsigned int u32x2 __attribute__((ext_vector_type(2)));

struct Pack128x2 { u32x4 lo, hi; };
struct Pack64x4  { u32x2 a, b, c, d; };

// Load a 16-element bf16 WMMA operand as two b128 LDS loads at dword
// offsets dwA / dwB within a row (row pointer must be 16B aligned).
__device__ __forceinline__ v16bf ld2x128(const __bf16* row, int dwA, int dwB) {
  const unsigned int* p = (const unsigned int*)row;
  Pack128x2 t;
  t.lo = *(const u32x4*)(p + dwA);
  t.hi = *(const u32x4*)(p + dwB);
  return __builtin_bit_cast(v16bf, t);
}
// Same but four b64 loads (for rows only 8B aligned), contiguous 8 dwords.
__device__ __forceinline__ v16bf ld4x64(const __bf16* row, int dw) {
  const unsigned int* p = (const unsigned int*)row;
  Pack64x4 t;
  t.a = *(const u32x2*)(p + dw + 0);
  t.b = *(const u32x2*)(p + dw + 2);
  t.c = *(const u32x2*)(p + dw + 4);
  t.d = *(const u32x2*)(p + dw + 6);
  return __builtin_bit_cast(v16bf, t);
}

__device__ __forceinline__ void atomicAddF32(float* p, float v) {
  unsafeAtomicAdd(p, v);   // native global_atomic_add_f32 on AMD
}

// ============================================================
// Phase A: flash attention per (b, h, 32-query tile), emitting only
// the column sums S[b,j] += sum_i heads[b,h,i,j] (fp32 atomics).
// 256 threads = 8 waves. Wave w owns output columns [128w, 128w+128).
// Each wave carries TWO 16-query score/accumulator sets so every staged
// K/V operand feeds 2 WMMAs (halves L2/fill traffic per FLOP).
// ============================================================
__global__ __launch_bounds__(256)
void attn_colsum_kernel(const float* __restrict__ x,
                        const float* __restrict__ Wq,
                        const float* __restrict__ Wk,
                        const float* __restrict__ Wv,
                        float* __restrict__ S)
{
  constexpr int QCS = 264;  // sQc row stride (bf16), 32 x 256 tile
  constexpr int KS  = 264;  // sK  row stride (bf16), 128 x 256 tile
  constexpr int VS  = 36;   // sVt row stride (bf16), 1024 x 32 tile (8B align, b64 loads)
  constexpr int PS  = 136;  // sP  row stride (bf16), 32 x 128 tile

  __shared__ __align__(16) __bf16 sQc[32 * QCS];     // 16896 B  (Q chunk)
  __shared__ __align__(16) __bf16 sP [32 * PS];      //  8704 B  (probabilities)
  __shared__ __align__(16) __bf16 sU [1024 * VS];    // 73728 B  (union: K tile / V^T tile)
  __shared__ float sM[32], sL[32], sFs[32], sLinv[32];
  __shared__ float sPmax[32][8], sPsum[32][8];

  const int tid  = threadIdx.x;
  const int lane = tid & 31;
  const int wv   = tid >> 5;        // wave 0..7
  const int g    = (lane >> 4);     // half of wave32 (WMMA layout)
  const int n16  = lane & 15;

  const int qt = blockIdx.x;        // 0..31 query tile (32 rows)
  const int h  = blockIdx.y;
  const int b  = blockIdx.z;
  const int q0 = qt * 32;

  const float* xb  = x  + (size_t)b * I_ * J_;
  const float* wqp = Wq + (size_t)h * I_ * J_;
  const float* wkp = Wk + (size_t)h * I_ * J_;
  const float* wvp = Wv + (size_t)h * I_ * J_;

  if (tid < 32) { sM[tid] = -3.0e38f; sL[tid] = 0.0f; }

  v8f acc[2][8];
  #pragma unroll
  for (int qs = 0; qs < 2; ++qs)
    #pragma unroll
    for (int s = 0; s < 8; ++s)
      acc[qs][s] = (v8f){0.f,0.f,0.f,0.f,0.f,0.f,0.f,0.f};

  const float inv32 = 0.03125f;     // 1/sqrt(J)

  for (int kc = 0; kc < 8; ++kc) {  // 128 keys per outer step
    const int k0 = kc * 128;

    // ---------- scores: this wave's 16-key subtile, full-J reduction ----------
    v8f sc[2];
    sc[0] = (v8f){0.f,0.f,0.f,0.f,0.f,0.f,0.f,0.f};
    sc[1] = (v8f){0.f,0.f,0.f,0.f,0.f,0.f,0.f,0.f};
    for (int jc = 0; jc < 4; ++jc) {
      const int jb = jc * 256;
      __syncthreads();              // sU free for reuse as K tile
      // fill sQc[32][256] = bf16(x * Wq) for this query tile / j-chunk
      {
        const int col = (tid & 63) * 4;
        #pragma unroll
        for (int rr = 0; rr < 8; ++rr) {
          const int row = (tid >> 6) + rr * 4;
          const float4 xv = *(const float4*)(xb  + (size_t)(q0 + row) * J_ + jb + col);
          const float4 wq4= *(const float4*)(wqp + (size_t)(q0 + row) * J_ + jb + col);
          bf16x4 o;
          o[0] = (__bf16)(xv.x * wq4.x); o[1] = (__bf16)(xv.y * wq4.y);
          o[2] = (__bf16)(xv.z * wq4.z); o[3] = (__bf16)(xv.w * wq4.w);
          *(bf16x4*)&sQc[row * QCS + col] = o;
        }
      }
      // fill sK[128][256] = bf16(x * Wk) for this key block / j-chunk
      {
        const int col = (tid & 63) * 4;
        for (int rr = 0; rr < 32; ++rr) {
          const int row = (tid >> 6) + rr * 4;
          const float4 xv = *(const float4*)(xb  + (size_t)(k0 + row) * J_ + jb + col);
          const float4 wk4= *(const float4*)(wkp + (size_t)(k0 + row) * J_ + jb + col);
          bf16x4 o;
          o[0] = (__bf16)(xv.x * wk4.x); o[1] = (__bf16)(xv.y * wk4.y);
          o[2] = (__bf16)(xv.z * wk4.z); o[3] = (__bf16)(xv.w * wk4.w);
          *(bf16x4*)&sU[row * KS + col] = o;
        }
      }
      __syncthreads();
      // 16 WMMAs over this 256-wide j chunk; B operand shared by both q-subtiles
      const __bf16* qrow0 = &sQc[n16 * QCS];                 // A: query rows 0-15
      const __bf16* qrow1 = &sQc[(16 + n16) * QCS];          // A: query rows 16-31
      const __bf16* krow  = &sU[(wv * 16 + n16) * KS];       // B: key row = wave subtile
      #pragma unroll
      for (int jc2 = 0; jc2 < 8; ++jc2) {
        const int dwb = jc2 * 16;
        v16bf bm = ld2x128(krow, dwb + 8 * g, dwb + 8 * g + 4);
        v16bf a0 = ld2x128(qrow0, dwb + 8 * g, dwb + 8 * g + 4);
        sc[0] = __builtin_amdgcn_wmma_f32_16x16x32_bf16(false, a0, false, bm,
                                                        (short)0, sc[0], false, false);
        v16bf a1 = ld2x128(qrow1, dwb + 8 * g, dwb + 8 * g + 4);
        sc[1] = __builtin_amdgcn_wmma_f32_16x16x32_bf16(false, a1, false, bm,
                                                        (short)0, sc[1], false, false);
      }
    }

    // ---------- online softmax across the 128 keys of this step ----------
    float lm[2][8];
    #pragma unroll
    for (int qs = 0; qs < 2; ++qs)
      #pragma unroll
      for (int r = 0; r < 8; ++r) {
        float v = sc[qs][r] * inv32;
        v = fmaxf(v, __shfl_xor(v, 1, 32));
        v = fmaxf(v, __shfl_xor(v, 2, 32));
        v = fmaxf(v, __shfl_xor(v, 4, 32));
        v = fmaxf(v, __shfl_xor(v, 8, 32));
        lm[qs][r] = v;                 // row max within this wave's 16 keys
      }
    if (n16 == 0) {
      #pragma unroll
      for (int qs = 0; qs < 2; ++qs)
        #pragma unroll
        for (int r = 0; r < 8; ++r) sPmax[qs * 16 + r + 8 * g][wv] = lm[qs][r];
    }
    __syncthreads();
    if (tid < 32) {
      float mo = sM[tid], mn = mo;
      #pragma unroll
      for (int w = 0; w < 8; ++w) mn = fmaxf(mn, sPmax[tid][w]);
      sFs[tid] = __expf(mo - mn);
      sM[tid]  = mn;
    }
    __syncthreads();
    // P = exp(s - m), write P tile, partial row sums, rescale accumulators
    float ps[2][8];
    #pragma unroll
    for (int qs = 0; qs < 2; ++qs)
      #pragma unroll
      for (int r = 0; r < 8; ++r) {
        const int row = qs * 16 + r + 8 * g;
        float p = __expf(sc[qs][r] * inv32 - sM[row]);
        sP[row * PS + wv * 16 + n16] = (__bf16)p;
        float t = p;
        t += __shfl_xor(t, 1, 32); t += __shfl_xor(t, 2, 32);
        t += __shfl_xor(t, 4, 32); t += __shfl_xor(t, 8, 32);
        ps[qs][r] = t;
      }
    if (n16 == 0) {
      #pragma unroll
      for (int qs = 0; qs < 2; ++qs)
        #pragma unroll
        for (int r = 0; r < 8; ++r) sPsum[qs * 16 + r + 8 * g][wv] = ps[qs][r];
    }
    #pragma unroll
    for (int qs = 0; qs < 2; ++qs) {
      float fs8[8];
      #pragma unroll
      for (int r = 0; r < 8; ++r) fs8[r] = sFs[qs * 16 + r + 8 * g];
      #pragma unroll
      for (int s = 0; s < 8; ++s)
        #pragma unroll
        for (int r = 0; r < 8; ++r) acc[qs][s][r] *= fs8[r];
    }
    __syncthreads();
    if (tid < 32) {
      float l = sL[tid] * sFs[tid];
      #pragma unroll
      for (int w = 0; w < 8; ++w) l += sPsum[tid][w];
      sL[tid] = l;
    }

    // ---------- P @ V over these 128 keys, V staged transposed ----------
    for (int vch = 0; vch < 4; ++vch) {
      __syncthreads();               // sU free (K tile / previous V tile done)
      const int kb = k0 + vch * 32;
      {
        const int keyg = tid >> 5;         // 8 keys per pass
        const int j4   = (tid & 31) * 4;
        #pragma unroll
        for (int kk = 0; kk < 4; ++kk) {
          const int krow = keyg + kk * 8;
          const float* xr = xb  + (size_t)(kb + krow) * J_;
          const float* wr = wvp + (size_t)(kb + krow) * J_;
          #pragma unroll
          for (int jj = 0; jj < 8; ++jj) {
            const int j = j4 + jj * 128;
            const float4 xv = *(const float4*)(xr + j);
            const float4 wv4= *(const float4*)(wr + j);
            sU[(j + 0) * VS + krow] = (__bf16)(xv.x * wv4.x);
            sU[(j + 1) * VS + krow] = (__bf16)(xv.y * wv4.y);
            sU[(j + 2) * VS + krow] = (__bf16)(xv.z * wv4.z);
            sU[(j + 3) * VS + krow] = (__bf16)(xv.w * wv4.w);
          }
        }
      }
      __syncthreads();
      const int dwb = vch * 16;
      v16bf a0 = ld2x128(&sP[n16 * PS],        dwb + 8 * g, dwb + 8 * g + 4);
      v16bf a1 = ld2x128(&sP[(16 + n16) * PS], dwb + 8 * g, dwb + 8 * g + 4);
      #pragma unroll
      for (int s = 0; s < 8; ++s) {
        const int ocol = wv * 128 + s * 16 + n16;  // B row = output column of V^T
        v16bf bm = ld4x64(&sU[ocol * VS], 8 * g);
        acc[0][s] = __builtin_amdgcn_wmma_f32_16x16x32_bf16(false, a0, false, bm,
                                                            (short)0, acc[0][s], false, false);
        acc[1][s] = __builtin_amdgcn_wmma_f32_16x16x32_bf16(false, a1, false, bm,
                                                            (short)0, acc[1][s], false, false);
      }
    }
  }

  // ---------- finalize: normalize by l, reduce rows, atomic into S ----------
  __syncthreads();
  if (tid < 32) sLinv[tid] = 1.0f / sL[tid];
  __syncthreads();
  float li[2][8];
  #pragma unroll
  for (int qs = 0; qs < 2; ++qs)
    #pragma unroll
    for (int r = 0; r < 8; ++r) li[qs][r] = sLinv[qs * 16 + r + 8 * g];
  #pragma unroll
  for (int s = 0; s < 8; ++s) {
    float cs = 0.f;
    #pragma unroll
    for (int qs = 0; qs < 2; ++qs)
      #pragma unroll
      for (int r = 0; r < 8; ++r) cs += acc[qs][s][r] * li[qs][r];
    cs += __shfl_xor(cs, 16, 32);    // merge rows 0-7 with rows 8-15 (same N)
    if (g == 0)
      atomicAddF32(&S[(size_t)b * J_ + wv * 128 + s * 16 + n16], cs);
  }
}

// ============================================================
// Phase B: attn_out = Wo[k,j]*S[b,j]; two LayerNorms over axis k
// with elementwise FFN in between. Streaming, recompute chain.
// grid (J/256, I/64, B), 256 threads.
// ============================================================
__global__ void zero_ws_kernel(float* p, int n) {
  int i = blockIdx.x * blockDim.x + threadIdx.x;
  if (i < n) p[i] = 0.f;
}

__global__ __launch_bounds__(256)
void ln1_stats_kernel(const float* __restrict__ x, const float* __restrict__ Wo,
                      const float* __restrict__ S, float* __restrict__ st1)
{
  const int j = blockIdx.x * 256 + threadIdx.x;
  const int kb = blockIdx.y * 64;
  const int b = blockIdx.z;
  const float sj = S[b * J_ + j];
  float sum = 0.f, sq = 0.f;
  for (int kk = 0; kk < 64; ++kk) {
    const int k = kb + kk;
    const float s0 = Wo[(size_t)k * J_ + j] * sj + x[((size_t)b * I_ + k) * J_ + j];
    sum += s0; sq += s0 * s0;
  }
  atomicAddF32(&st1[(b * J_ + j) * 2 + 0], sum);
  atomicAddF32(&st1[(b * J_ + j) * 2 + 1], sq);
}

__device__ __forceinline__ float chain_z(float s0, float mu, float rs,
                                         float g1k, float b1k,
                                         float x1e, float y1e, float x2e, float y2e)
{
  const float s1 = (s0 - mu) * rs * g1k + b1k;
  const float ff = fmaxf(s1 * x1e + y1e, 0.f);
  return ff * x2e + y2e + s1;
}

__global__ __launch_bounds__(256)
void ln2_stats_kernel(const float* __restrict__ x, const float* __restrict__ Wo,
                      const float* __restrict__ S, const float* __restrict__ st1,
                      const float* __restrict__ x1, const float* __restrict__ y1,
                      const float* __restrict__ x2, const float* __restrict__ y2,
                      const float* __restrict__ g1, const float* __restrict__ b1,
                      float* __restrict__ st2)
{
  const int j = blockIdx.x * 256 + threadIdx.x;
  const int kb = blockIdx.y * 64;
  const int b = blockIdx.z;
  const float sj = S[b * J_ + j];
  const float mu = st1[(b * J_ + j) * 2 + 0] * (1.f / I_);
  const float vr = st1[(b * J_ + j) * 2 + 1] * (1.f / I_) - mu * mu;
  const float rs = rsqrtf(vr + EPSLN);
  float sum = 0.f, sq = 0.f;
  for (int kk = 0; kk < 64; ++kk) {
    const int k = kb + kk;
    const size_t e = (size_t)k * J_ + j;
    const float s0 = Wo[e] * sj + x[((size_t)b * I_ + k) * J_ + j];
    const float z = chain_z(s0, mu, rs, g1[k], b1[k], x1[e], y1[e], x2[e], y2[e]);
    sum += z; sq += z * z;
  }
  atomicAddF32(&st2[(b * J_ + j) * 2 + 0], sum);
  atomicAddF32(&st2[(b * J_ + j) * 2 + 1], sq);
}

__global__ __launch_bounds__(256)
void final_kernel(const float* __restrict__ x, const float* __restrict__ Wo,
                  const float* __restrict__ S, const float* __restrict__ st1,
                  const float* __restrict__ st2,
                  const float* __restrict__ x1, const float* __restrict__ y1,
                  const float* __restrict__ x2, const float* __restrict__ y2,
                  const float* __restrict__ g1, const float* __restrict__ b1,
                  const float* __restrict__ g2, const float* __restrict__ b2,
                  float* __restrict__ out)
{
  const int j = blockIdx.x * 256 + threadIdx.x;
  const int kb = blockIdx.y * 64;
  const int b = blockIdx.z;
  const float sj = S[b * J_ + j];
  const float mu = st1[(b * J_ + j) * 2 + 0] * (1.f / I_);
  const float vr = st1[(b * J_ + j) * 2 + 1] * (1.f / I_) - mu * mu;
  const float rs = rsqrtf(vr + EPSLN);
  const float mu2 = st2[(b * J_ + j) * 2 + 0] * (1.f / I_);
  const float vr2 = st2[(b * J_ + j) * 2 + 1] * (1.f / I_) - mu2 * mu2;
  const float rs2 = rsqrtf(vr2 + EPSLN);
  for (int kk = 0; kk < 64; ++kk) {
    const int k = kb + kk;
    const size_t e = (size_t)k * J_ + j;
    const float s0 = Wo[e] * sj + x[((size_t)b * I_ + k) * J_ + j];
    const float z = chain_z(s0, mu, rs, g1[k], b1[k], x1[e], y1[e], x2[e], y2[e]);
    out[((size_t)b * I_ + k) * J_ + j] = (z - mu2) * rs2 * g2[k] + b2[k];
  }
}

// ============================================================
extern "C" void kernel_launch(void* const* d_in, const int* in_sizes, int n_in,
                              void* d_out, int out_size, void* d_ws, size_t ws_size,
                              hipStream_t stream) {
  const float* x  = (const float*)d_in[0];
  const float* Wq = (const float*)d_in[1];
  const float* Wk = (const float*)d_in[2];
  const float* Wv = (const float*)d_in[3];
  const float* Wo = (const float*)d_in[4];
  const float* x1 = (const float*)d_in[5];
  const float* y1 = (const float*)d_in[6];
  const float* x2 = (const float*)d_in[7];
  const float* y2 = (const float*)d_in[8];
  const float* g1 = (const float*)d_in[9];
  const float* b1 = (const float*)d_in[10];
  const float* g2 = (const float*)d_in[11];
  const float* b2 = (const float*)d_in[12];
  float* out = (float*)d_out;

  float* S   = (float*)d_ws;                 // [B, J]
  float* st1 = S + B_ * J_;                  // [B, J, 2]
  float* st2 = st1 + 2 * B_ * J_;            // [B, J, 2]
  const int nzero = 5 * B_ * J_;

  zero_ws_kernel<<<(nzero + 255) / 256, 256, 0, stream>>>((float*)d_ws, nzero);
  attn_colsum_kernel<<<dim3(I_ / 32, H_, B_), 256, 0, stream>>>(x, Wq, Wk, Wv, S);
  ln1_stats_kernel<<<dim3(J_ / 256, I_ / 64, B_), 256, 0, stream>>>(x, Wo, S, st1);
  ln2_stats_kernel<<<dim3(J_ / 256, I_ / 64, B_), 256, 0, stream>>>(
      x, Wo, S, st1, x1, y1, x2, y2, g1, b1, st2);
  final_kernel<<<dim3(J_ / 256, I_ / 64, B_), 256, 0, stream>>>(
      x, Wo, S, st1, st2, x1, y1, x2, y2, g1, b1, g2, b2, out);
  (void)in_sizes; (void)n_in; (void)out_size; (void)ws_size;
}
